// YOLOV4Head_2267742732658
// MI455X (gfx1250) — compile-verified, matrix operand order checked by reading, not run on hardware
//
#include <hip/hip_runtime.h>
#include <hip/hip_bf16.h>
#include <stdint.h>

// ---------------------------------------------------------------------------
// YOLOv4 head for MI455X (gfx1250): bf16 implicit-GEMM convs on WMMA,
// async global->LDS tile staging (ASYNCcnt), fused BN/LeakyReLU epilogue,
// fused decode+score-threshold, block NMS.
// ---------------------------------------------------------------------------

typedef __attribute__((ext_vector_type(16))) __bf16 v16bf;
typedef __attribute__((ext_vector_type(8)))  float  v8f;

#define NCLS       80
#define PRE_NMS_K  400
#define MAX_DET_K  100
#define CAND_MAX   16384
#define A_TOTAL    22743      // 3*(76^2 + 38^2 + 19^2)
#define IMG2       1216.0f    // 2*IMG_SIZE class offset

__device__ __forceinline__ __bf16 f2bf(float f) {
  uint32_t u = __builtin_bit_cast(uint32_t, f);
  u += 0x7FFFu + ((u >> 16) & 1u);          // round-to-nearest-even
  uint16_t h = (uint16_t)(u >> 16);
  return __builtin_bit_cast(__bf16, h);
}

// pack two f32 -> two bf16 (RNE) in one dword
__device__ __forceinline__ uint32_t pack2bf(float a, float b) {
  uint32_t ua = __builtin_bit_cast(uint32_t, a);
  ua += 0x7FFFu + ((ua >> 16) & 1u);
  uint32_t ub = __builtin_bit_cast(uint32_t, b);
  ub += 0x7FFFu + ((ub >> 16) & 1u);
  return (ua >> 16) | (ub & 0xFFFF0000u);
}

// 32-bit LDS byte address of a __shared__ object (generic -> addrspace(3))
__device__ __forceinline__ uint32_t lds_addr_of(const void* p) {
  return (uint32_t)(uintptr_t)(__attribute__((address_space(3))) const void*)p;
}

// async global->LDS 16B copy, tracked by ASYNCcnt
__device__ __forceinline__ void async_copy_b128(uint32_t lds_byte_addr,
                                                const void* gptr) {
  uint64_t ga = (uint64_t)(uintptr_t)gptr;
  asm volatile("global_load_async_to_lds_b128 %0, %1, off"
               :: "v"(lds_byte_addr), "v"(ga) : "memory");
}

__device__ __forceinline__ void async_wait0() {
  asm volatile("s_wait_asynccnt 0x0" ::: "memory");
}

__device__ __forceinline__ v16bf frag_load(const __bf16* p) {
  union { uint4 u[2]; v16bf v; } x;
  x.u[0] = *(const uint4*)(p);        // K 0..7   (or 8..15 for hi lanes)
  x.u[1] = *(const uint4*)(p + 16);   // K 16..23 (or 24..31 for hi lanes)
  return x.v;
}

// ---- weight convert + transpose: w[k][n] (f32) -> wt[n][k] (bf16) ----------
__global__ void prep_wT_kernel(const float* __restrict__ w,
                               __bf16* __restrict__ wt,
                               int K, int N, int total) {
  int i = blockIdx.x * blockDim.x + threadIdx.x;
  if (i >= total) return;
  int k = i / N, n = i % N;
  wt[(size_t)n * K + k] = f2bf(w[(size_t)k * N + n]);
}

// ---- 3x3 conv as implicit GEMM + BN scale/bias + LeakyReLU -----------------
// A: im2col of NHWC feature (M x K), B: wt [N][K] bf16, C: act [M][HD] bf16
__global__ __launch_bounds__(256) void conv3x3_wmma_kernel(
    const float* __restrict__ f, const __bf16* __restrict__ w3t,
    const float* __restrict__ gamma, const float* __restrict__ beta,
    __bf16* __restrict__ act,
    int H, int Cin, int HD, int M, int K) {
  constexpr int AST = 40, BST = 40;   // padded LDS strides (80B rows, 16B aligned)
  __shared__ __bf16 Alds[128 * AST];
  __shared__ __bf16 Blds[128 * BST];

  const int tid  = threadIdx.x;
  const int lane = tid & 31;
  const int wave = tid >> 5;
  const int m0 = blockIdx.x * 128;
  const int n0 = blockIdx.y * 128;
  const int wm = (wave & 3) * 32;     // wave row base: 2 frags of 16
  const int wn = (wave >> 2) * 64;    // wave col base: 4 frags of 16

  v8f acc[2][4];
#pragma unroll
  for (int i = 0; i < 2; i++)
#pragma unroll
    for (int j = 0; j < 4; j++) { v8f z = {}; acc[i][j] = z; }

  const int row_l = tid >> 1;         // staging row 0..127
  const int half  = tid & 1;          // 16-element half of the 32-wide K slab
  const int HW = H * H;
  const int m = m0 + row_l;
  int b = 0, y = 0, x = 0;
  const bool mrow_ok = (m < M);
  if (mrow_ok) { b = m / HW; int r = m % HW; y = r / H; x = r % H; }

  const int lm = lane & 15;
  const int kb = (lane >> 4) * 8;
  const uint32_t b_lds = lds_addr_of(&Blds[row_l * BST + half * 16]);

  for (int k0 = 0; k0 < K; k0 += 32) {
    // --- stage A: im2col 128x32, f32 -> bf16 packed dword stores ---
    const int kykx = k0 / Cin;
    const int ci0  = k0 % Cin;
    const int ky = kykx / 3, kx = kykx % 3;
    const int yy = y + ky - 1, xx = x + kx - 1;
    const bool inb = mrow_ok && (yy >= 0) && (yy < H) && (xx >= 0) && (xx < H);
    const float* src = f + (((size_t)(b * H + yy) * H + xx) * Cin + ci0 + half * 16);
    uint2* adst2 = (uint2*)&Alds[row_l * AST + half * 16];
#pragma unroll
    for (int q = 0; q < 4; q++) {
      float4 v = inb ? *(const float4*)(src + q * 4) : make_float4(0.f, 0.f, 0.f, 0.f);
      uint2 pv; pv.x = pack2bf(v.x, v.y); pv.y = pack2bf(v.z, v.w);
      adst2[q] = pv;
    }
    // --- stage B: async global->LDS, bf16 weights contiguous along K ---
    {
      const __bf16* bsrc = w3t + (size_t)(n0 + row_l) * K + k0 + half * 16;
      async_copy_b128(b_lds, bsrc);
      async_copy_b128(b_lds + 16, bsrc + 8);
    }
    if (inb && (k0 + 32 < K)) __builtin_prefetch((const void*)(src + 32), 0, 0);
    async_wait0();
    __syncthreads();

    // --- compute: 2x4 fragments of v_wmma_f32_16x16x32_bf16 ---
    v16bf afrag[2], bfrag[4];
#pragma unroll
    for (int i = 0; i < 2; i++)
      afrag[i] = frag_load(&Alds[(wm + i * 16 + lm) * AST + kb]);
#pragma unroll
    for (int j = 0; j < 4; j++)
      bfrag[j] = frag_load(&Blds[(wn + j * 16 + lm) * BST + kb]);
#pragma unroll
    for (int i = 0; i < 2; i++)
#pragma unroll
      for (int j = 0; j < 4; j++)
        acc[i][j] = __builtin_amdgcn_wmma_f32_16x16x32_bf16(
            false, afrag[i], false, bfrag[j], (short)0, acc[i][j], false, false);
    __syncthreads();
  }

  // --- epilogue: gamma/beta + LeakyReLU(0.1), store bf16 activations ---
  const int hi8 = (lane >> 4) << 3;
#pragma unroll
  for (int i = 0; i < 2; i++) {
#pragma unroll
    for (int j = 0; j < 4; j++) {
      const int col = n0 + wn + j * 16 + lm;
      const float g = gamma[col], bt = beta[col];
#pragma unroll
      for (int r = 0; r < 8; r++) {
        const int row = m0 + wm + i * 16 + hi8 + r;
        if (row < M) {
          float v = acc[i][j][r] * g + bt;
          v = v > 0.0f ? v : 0.1f * v;
          act[(size_t)row * HD + col] = f2bf(v);
        }
      }
    }
  }
}

// ---- 1x1 conv GEMM: act(M x HD) x w1t(255 x HD)^T + b1 -> pred f32 --------
__global__ __launch_bounds__(256) void conv1x1_wmma_kernel(
    const __bf16* __restrict__ actIn, const __bf16* __restrict__ w1t,
    const float* __restrict__ b1, float* __restrict__ pred,
    int K /*=HD*/, int M) {
  constexpr int AST = 40, BST = 40;
  __shared__ __bf16 Alds[128 * AST];
  __shared__ __bf16 Blds[128 * BST];

  const int tid  = threadIdx.x;
  const int lane = tid & 31;
  const int wave = tid >> 5;
  const int m0 = blockIdx.x * 128;
  const int n0 = blockIdx.y * 128;
  const int wm = (wave & 3) * 32;
  const int wn = (wave >> 2) * 64;

  v8f acc[2][4];
#pragma unroll
  for (int i = 0; i < 2; i++)
#pragma unroll
    for (int j = 0; j < 4; j++) { v8f z = {}; acc[i][j] = z; }

  const int row_l = tid >> 1;
  const int half  = tid & 1;
  const int lm = lane & 15;
  const int kb = (lane >> 4) * 8;
  const uint4 z4 = make_uint4(0u, 0u, 0u, 0u);
  const bool a_ok = (m0 + row_l < M);
  const bool b_ok = (n0 + row_l < 255);
  const uint32_t a_lds = lds_addr_of(&Alds[row_l * AST + half * 16]);
  const uint32_t b_lds = lds_addr_of(&Blds[row_l * BST + half * 16]);

  for (int k0 = 0; k0 < K; k0 += 32) {
    if (a_ok) { // async stage A from bf16 activations
      const __bf16* asrc = actIn + (size_t)(m0 + row_l) * K + k0 + half * 16;
      async_copy_b128(a_lds, asrc);
      async_copy_b128(a_lds + 16, asrc + 8);
    } else {
      uint4* adst = (uint4*)&Alds[row_l * AST + half * 16];
      adst[0] = z4; adst[1] = z4;
    }
    if (b_ok) { // async stage B from transposed bf16 1x1 weights
      const __bf16* bsrc = w1t + (size_t)(n0 + row_l) * K + k0 + half * 16;
      async_copy_b128(b_lds, bsrc);
      async_copy_b128(b_lds + 16, bsrc + 8);
    } else {
      uint4* bdst = (uint4*)&Blds[row_l * BST + half * 16];
      bdst[0] = z4; bdst[1] = z4;
    }
    async_wait0();
    __syncthreads();

    v16bf afrag[2], bfrag[4];
#pragma unroll
    for (int i = 0; i < 2; i++)
      afrag[i] = frag_load(&Alds[(wm + i * 16 + lm) * AST + kb]);
#pragma unroll
    for (int j = 0; j < 4; j++)
      bfrag[j] = frag_load(&Blds[(wn + j * 16 + lm) * BST + kb]);
#pragma unroll
    for (int i = 0; i < 2; i++)
#pragma unroll
      for (int j = 0; j < 4; j++)
        acc[i][j] = __builtin_amdgcn_wmma_f32_16x16x32_bf16(
            false, afrag[i], false, bfrag[j], (short)0, acc[i][j], false, false);
    __syncthreads();
  }

  const int hi8 = (lane >> 4) << 3;
#pragma unroll
  for (int i = 0; i < 2; i++) {
#pragma unroll
    for (int j = 0; j < 4; j++) {
      const int col = n0 + wn + j * 16 + lm;
      if (col < 255) {
        const float bb = b1[col];
#pragma unroll
        for (int r = 0; r < 8; r++) {
          const int row = m0 + wm + i * 16 + hi8 + r;
          if (row < M) pred[(size_t)row * 255 + col] = acc[i][j][r] + bb;
        }
      }
    }
  }
}

// ---- decode: boxes + fused score-threshold candidate push ------------------
__global__ void decode_kernel(const float* __restrict__ pred,
                              float* __restrict__ boxes,
                              float* __restrict__ cand_val,
                              int* __restrict__ cand_idx,
                              int* __restrict__ cand_cnt,
                              int H, float stride, int lvl_off,
                              float aw0, float ah0, float aw1, float ah1,
                              float aw2, float ah2) {
  int gid = blockIdx.x * blockDim.x + threadIdx.x;
  int total = 8 * H * H * 3;
  if (gid >= total) return;
  int a = gid % 3; int t = gid / 3;
  int x = t % H; t /= H;
  int y = t % H; int b = t / H;
  const float* p = pred + (size_t)((b * H + y) * H + x) * 255 + a * 85;
  float sx = 1.f / (1.f + __expf(-p[0]));
  float sy = 1.f / (1.f + __expf(-p[1]));
  float cx = (sx * 1.05f - 0.025f + (float)x) * stride;
  float cy = (sy * 1.05f - 0.025f + (float)y) * stride;
  float aw = (a == 0) ? aw0 : ((a == 1) ? aw1 : aw2);
  float ah = (a == 0) ? ah0 : ((a == 1) ? ah1 : ah2);
  float w = __expf(p[2]) * aw, h = __expf(p[3]) * ah;
  int anchor = lvl_off + (y * H + x) * 3 + a;
  float* bx = boxes + ((size_t)b * A_TOTAL + anchor) * 4;
  bx[0] = cx - 0.5f * w; bx[1] = cy - 0.5f * h;
  bx[2] = cx + 0.5f * w; bx[3] = cy + 0.5f * h;
  float so = 1.f / (1.f + __expf(-p[4]));
  for (int c = 0; c < NCLS; c++) {
    float sc = so * (1.f / (1.f + __expf(-p[5 + c])));
    if (sc >= 0.05f) {
      int pos = atomicAdd(&cand_cnt[b], 1);
      if (pos < CAND_MAX) {
        cand_val[b * CAND_MAX + pos] = sc;
        cand_idx[b * CAND_MAX + pos] = anchor * NCLS + c;
      }
    }
  }
}

// ---- NMS: one block per image ---------------------------------------------
__global__ __launch_bounds__(256) void nms_kernel(
    const float* __restrict__ boxes, float* __restrict__ cand_val,
    const int* __restrict__ cand_idx, const int* __restrict__ cand_cnt,
    float* __restrict__ out) {
  __shared__ float s_val[PRE_NMS_K];
  __shared__ int   s_idx[PRE_NMS_K];
  __shared__ float s_box[PRE_NMS_K * 4];   // class-offset boxes
  __shared__ int   s_keep[PRE_NMS_K];
  __shared__ float r_val[256];
  __shared__ int   r_idx[256];

  const int b = blockIdx.x;
  const int tid = threadIdx.x;
  int cnt = cand_cnt[b]; if (cnt > CAND_MAX) cnt = CAND_MAX;
  float* cv = cand_val + b * CAND_MAX;
  const int* ci = cand_idx + b * CAND_MAX;

  // --- top-PRE_NMS by destructive block argmax ---
  for (int it = 0; it < PRE_NMS_K; it++) {
    float best = 0.0f; int besti = -1;
    for (int j = tid; j < cnt; j += 256) {
      float v = cv[j];
      if (v > best) { best = v; besti = j; }
    }
    r_val[tid] = best; r_idx[tid] = besti;
    __syncthreads();
    for (int s = 128; s > 0; s >>= 1) {
      if (tid < s && r_val[tid + s] > r_val[tid]) {
        r_val[tid] = r_val[tid + s]; r_idx[tid] = r_idx[tid + s];
      }
      __syncthreads();
    }
    if (tid == 0) {
      int bi = r_idx[0];
      if (bi >= 0 && r_val[0] > 0.0f) {
        s_val[it] = r_val[0]; s_idx[it] = ci[bi]; cv[bi] = -1.0f;
      } else { s_val[it] = 0.0f; s_idx[it] = 0; }
    }
    __syncthreads();
  }

  // --- build class-offset boxes ---
  for (int i = tid; i < PRE_NMS_K; i += 256) {
    int fi = s_idx[i];
    int a = fi / NCLS, c = fi % NCLS;
    const float* bx = boxes + ((size_t)b * A_TOTAL + a) * 4;
    float off = (float)c * IMG2;
    s_box[i * 4 + 0] = bx[0] + off; s_box[i * 4 + 1] = bx[1] + off;
    s_box[i * 4 + 2] = bx[2] + off; s_box[i * 4 + 3] = bx[3] + off;
    s_keep[i] = 1;
  }
  __syncthreads();

  // --- sequential greedy suppression, parallel inner IoU ---
  for (int i = 0; i < PRE_NMS_K; i++) {
    bool active = s_keep[i] && (s_val[i] > 0.0f);
    if (active) {
      float ax1 = s_box[i * 4 + 0], ay1 = s_box[i * 4 + 1];
      float ax2 = s_box[i * 4 + 2], ay2 = s_box[i * 4 + 3];
      float areaA = (ax2 - ax1) * (ay2 - ay1);
      for (int j = i + 1 + tid; j < PRE_NMS_K; j += 256) {
        if (s_keep[j]) {
          float bx1 = s_box[j * 4 + 0], by1 = s_box[j * 4 + 1];
          float bx2 = s_box[j * 4 + 2], by2 = s_box[j * 4 + 3];
          float lx = fmaxf(ax1, bx1), ly = fmaxf(ay1, by1);
          float rx = fminf(ax2, bx2), ry = fminf(ay2, by2);
          float iw = fmaxf(rx - lx, 0.f), ih = fmaxf(ry - ly, 0.f);
          float inter = iw * ih;
          float areaB = (bx2 - bx1) * (by2 - by1);
          float iou = inter / (areaA + areaB - inter + 1e-7f);
          if (iou > 0.5f) s_keep[j] = 0;
        }
      }
    }
    __syncthreads();
  }

  for (int i = tid; i < PRE_NMS_K; i += 256)
    if (!s_keep[i]) s_val[i] = 0.0f;
  __syncthreads();

  // --- final top-MAX_DET ---
  float* ob = out;                         // (8,100,4)
  float* os = out + 8 * MAX_DET_K * 4;     // (8,100)
  float* oc = os + 8 * MAX_DET_K;          // (8,100) classes as float
  for (int it = 0; it < MAX_DET_K; it++) {
    float best = -1.0f; int besti = 0;
    for (int j = tid; j < PRE_NMS_K; j += 256) {
      float v = s_val[j];
      if (v > best) { best = v; besti = j; }
    }
    r_val[tid] = best; r_idx[tid] = besti;
    __syncthreads();
    for (int s = 128; s > 0; s >>= 1) {
      if (tid < s && r_val[tid + s] > r_val[tid]) {
        r_val[tid] = r_val[tid + s]; r_idx[tid] = r_idx[tid + s];
      }
      __syncthreads();
    }
    if (tid == 0) {
      int bi = r_idx[0];
      float v = r_val[0] > 0.f ? r_val[0] : 0.f;
      int c = s_idx[bi] % NCLS;
      float off = (float)c * IMG2;
      ob[(b * MAX_DET_K + it) * 4 + 0] = s_box[bi * 4 + 0] - off;
      ob[(b * MAX_DET_K + it) * 4 + 1] = s_box[bi * 4 + 1] - off;
      ob[(b * MAX_DET_K + it) * 4 + 2] = s_box[bi * 4 + 2] - off;
      ob[(b * MAX_DET_K + it) * 4 + 3] = s_box[bi * 4 + 3] - off;
      os[b * MAX_DET_K + it] = v;
      oc[b * MAX_DET_K + it] = (float)c;
      s_val[bi] = -1.0f;
    }
    __syncthreads();
  }
}

__global__ void init_kernel(int* cand_cnt) {
  if (threadIdx.x < 8 && blockIdx.x == 0) cand_cnt[threadIdx.x] = 0;
}

// ---------------------------------------------------------------------------
extern "C" void kernel_launch(void* const* d_in, const int* in_sizes, int n_in,
                              void* d_out, int out_size, void* d_ws, size_t ws_size,
                              hipStream_t stream) {
  (void)in_sizes; (void)n_in; (void)out_size; (void)ws_size;

  const int   Hh[3]  = {76, 38, 19};
  const int   CIN[3] = {128, 256, 512};
  const int   HDv[3] = {256, 512, 1024};
  const int   K3[3]  = {1152, 2304, 4608};
  const int   Mv[3]  = {46208, 11552, 2888};
  const float STR[3] = {8.f, 16.f, 32.f};
  const int   LOFF[3] = {0, 3 * 5776, 3 * 5776 + 3 * 1444};
  const float AW[3][3] = {{12.f, 19.f, 40.f}, {36.f, 76.f, 72.f}, {142.f, 192.f, 459.f}};
  const float AH[3][3] = {{16.f, 36.f, 28.f}, {75.f, 55.f, 146.f}, {110.f, 243.f, 401.f}};

  // inputs: per level i: f, w3, gamma, beta, w1, b1 at indices 6*i + {0..5}
  const float* F[3]; const float* W3[3]; const float* G[3];
  const float* BT[3]; const float* W1[3]; const float* B1[3];
  for (int i = 0; i < 3; i++) {
    F[i]  = (const float*)d_in[6 * i + 0];
    W3[i] = (const float*)d_in[6 * i + 1];
    G[i]  = (const float*)d_in[6 * i + 2];
    BT[i] = (const float*)d_in[6 * i + 3];
    W1[i] = (const float*)d_in[6 * i + 4];
    B1[i] = (const float*)d_in[6 * i + 5];
  }

  // workspace carve-up
  unsigned char* p = (unsigned char*)d_ws;
  auto take = [&](size_t bytes) {
    unsigned char* r = p;
    p += (bytes + 255) & ~(size_t)255;
    return r;
  };
  __bf16* w3t[3]; __bf16* w1t[3]; __bf16* act[3]; float* pred[3];
  for (int i = 0; i < 3; i++) w3t[i] = (__bf16*)take((size_t)K3[i] * HDv[i] * 2);
  for (int i = 0; i < 3; i++) w1t[i] = (__bf16*)take((size_t)255 * HDv[i] * 2);
  for (int i = 0; i < 3; i++) act[i] = (__bf16*)take((size_t)Mv[i] * HDv[i] * 2);
  for (int i = 0; i < 3; i++) pred[i] = (float*)take((size_t)Mv[i] * 255 * 4);
  float* boxes    = (float*)take((size_t)8 * A_TOTAL * 4 * 4);
  float* cand_val = (float*)take((size_t)8 * CAND_MAX * 4);
  int*   cand_idx = (int*)take((size_t)8 * CAND_MAX * 4);
  int*   cand_cnt = (int*)take((size_t)8 * 4);
  float* out = (float*)d_out;

  init_kernel<<<1, 256, 0, stream>>>(cand_cnt);

  for (int i = 0; i < 3; i++) {
    {
      int tot = K3[i] * HDv[i];
      prep_wT_kernel<<<(tot + 255) / 256, 256, 0, stream>>>(W3[i], w3t[i], K3[i], HDv[i], tot);
    }
    {
      int tot = HDv[i] * 255;
      prep_wT_kernel<<<(tot + 255) / 256, 256, 0, stream>>>(W1[i], w1t[i], HDv[i], 255, tot);
    }
    {
      dim3 grid((Mv[i] + 127) / 128, HDv[i] / 128);
      conv3x3_wmma_kernel<<<grid, 256, 0, stream>>>(
          F[i], w3t[i], G[i], BT[i], act[i], Hh[i], CIN[i], HDv[i], Mv[i], K3[i]);
    }
    {
      dim3 grid((Mv[i] + 127) / 128, 2);
      conv1x1_wmma_kernel<<<grid, 256, 0, stream>>>(
          act[i], w1t[i], B1[i], pred[i], HDv[i], Mv[i]);
    }
    {
      int tot = 8 * Hh[i] * Hh[i] * 3;
      decode_kernel<<<(tot + 255) / 256, 256, 0, stream>>>(
          pred[i], boxes, cand_val, cand_idx, cand_cnt, Hh[i], STR[i], LOFF[i],
          AW[i][0], AH[i][0], AW[i][1], AH[i][1], AW[i][2], AH[i][2]);
    }
  }

  nms_kernel<<<8, 256, 0, stream>>>(boxes, cand_val, cand_idx, cand_cnt, out);
}
